// AMHSA_22771916604108
// MI455X (gfx1250) — matrix operand order repaired and checked by reading, not compiled, not run on hardware
//
#include <hip/hip_runtime.h>
#include <hip/hip_bf16.h>

// ---------------------------------------------------------------------------
// CDNA5 (gfx1250) implementation.
// All GEMMs and both attention matmuls use v_wmma_f32_16x16x32_f16 (f16 in,
// f32 accumulate). Operands are pre-packed into fragment-native f16 layouts
// so hot loops are pure b128 loads + WMMA (no branches, no converts). The
// GEMM K-loop is fully unrolled (KT=8 template) so the scheduler can overlap
// global_load_b128 issue with the WMMA pipe instead of draining loadcnt to 0.
// Attention is transposed flash attention (softmax stats per lane, wave32).
// ---------------------------------------------------------------------------

typedef __attribute__((ext_vector_type(16))) _Float16 v16h;
typedef __attribute__((ext_vector_type(8)))  _Float16 v8h;
typedef __attribute__((ext_vector_type(8)))  float    v8f;

union H16 { v16h v; v8h h[2]; };

#define WMMA_F16(a, b, c) \
  __builtin_amdgcn_wmma_f32_16x16x32_f16(false, (a), false, (b), (short)0, (c), false, false)

// ---------------------------------------------------------------------------
// Pack weight W[m, k] (index m*w_rs + k*w_cs) -> Wp[kt][Mp][32] f16, zero-pad.
// ---------------------------------------------------------------------------
__global__ void pack_w_kernel(const float* __restrict__ W, int w_rs, int w_cs,
                              int M, int K, int Mp, int Kt,
                              _Float16* __restrict__ Wp)
{
  int idx = blockIdx.x * blockDim.x + threadIdx.x;
  if (idx >= Kt * Mp * 32) return;
  int kl  = idx & 31;
  int row = (idx >> 5) % Mp;
  int kt  = (idx >> 5) / Mp;
  int k   = kt * 32 + kl;
  float v = (row < M && k < K) ? W[(size_t)row * w_rs + (size_t)k * w_cs] : 0.0f;
  Wp[idx] = (_Float16)v;
}

// ---------------------------------------------------------------------------
// Pack activation X[k, n] (batched) -> Xp[kt][n][32] f16, zero-pad K.
// grid.z = batch
// ---------------------------------------------------------------------------
__global__ void pack_x_kernel(const float* __restrict__ X, long xbs,
                              int K, int N, int Kt,
                              _Float16* __restrict__ Xp, long pbs)
{
  int idx = blockIdx.x * blockDim.x + threadIdx.x;
  if (idx >= Kt * N * 32) return;
  int b  = blockIdx.z;
  int kl = idx & 31;
  int n  = (idx >> 5) % N;
  int kt = (idx >> 5) / N;
  int k  = kt * 32 + kl;
  float v = (k < K) ? X[(long)b * xbs + (size_t)k * N + n] : 0.0f;
  Xp[(long)b * pbs + idx] = (_Float16)v;
}

// ---------------------------------------------------------------------------
// Pack qkv f32 [b][768][4096] into attention-friendly f16 buffers:
//   Qp[bh][n][64], Kp[bh][m][64]  (transposed), Vp[bh][d][4096] (converted)
// ---------------------------------------------------------------------------
__global__ void pack_qkv_kernel(const float* __restrict__ qkv,
                                _Float16* __restrict__ Qp,
                                _Float16* __restrict__ Kp,
                                _Float16* __restrict__ Vp)
{
  int idx = blockIdx.x * blockDim.x + threadIdx.x;
  if (idx >= 2 * 4 * 4096 * 64) return;
  int dh = idx & 63;
  int n  = (idx >> 6) & 4095;
  int h  = (idx >> 18) & 3;
  int b  = idx >> 20;
  const float* base = qkv + (size_t)b * 768 * 4096;
  float qv = base[(size_t)(      h * 64 + dh) * 4096 + n];
  float kv = base[(size_t)(256 + h * 64 + dh) * 4096 + n];
  float vv = base[(size_t)(512 + h * 64 + dh) * 4096 + n];
  size_t t = ((size_t)(b * 4 + h) * 4096 + n) * 64 + dh;      // transposed
  Qp[t] = (_Float16)qv;
  Kp[t] = (_Float16)kv;
  Vp[((size_t)(b * 4 + h) * 64 + dh) * 4096 + n] = (_Float16)vv;
}

// ---------------------------------------------------------------------------
// Packed GEMM: C[m,n] (+)= sum_k Wp * Xp  (+ bias[m])
//   Wp: [KT][Mp][32] f16, Xp: [KT][N][32] f16 (batch stride pbs halves)
//   mode 0: C = v ; 1: C += v ; 2: C = R + scale*v
// One wave per 16x16 tile. K-loop fully unrolled (KT compile-time).
// ---------------------------------------------------------------------------
template <int KT>
__global__ __launch_bounds__(256) void gemm_wmma_packed(
    const _Float16* __restrict__ Wp,
    const _Float16* __restrict__ Xp, long pbs,
    const float* __restrict__ bias,
    float* C, long cbs,
    const float* __restrict__ R, long rbs, float scale,
    int Mp, int Mout, int N, int mode)
{
  const int lane = threadIdx.x & 31;
  const int wave = threadIdx.x >> 5;
  const int tiles_n = N >> 4;
  const int tiles_m = Mp >> 4;
  const long tile = (long)blockIdx.x * 8 + wave;
  if (tile >= (long)tiles_m * tiles_n) return;
  const int mt = (int)(tile / tiles_n);
  const int nt = (int)(tile % tiles_n);
  const int b  = blockIdx.z;

  const int l16 = lane & 15;
  const int hh  = lane >> 4;
  const int ma  = mt * 16 + l16;
  const int nb  = nt * 16 + l16;

  const v8h* __restrict__ wa = (const v8h*)(Wp + (size_t)ma * 32);
  const v8h* __restrict__ xa = (const v8h*)(Xp + (long)b * pbs + (size_t)nb * 32);
  const size_t wstep = (size_t)Mp * 4;   // v8h units per kt
  const size_t xstep = (size_t)N  * 4;

  // deep speculative prefetch into the K panel (no branches)
  __builtin_prefetch(wa + 4 * wstep, 0, 1);
  __builtin_prefetch(xa + 4 * xstep, 0, 1);

  v8f acc = {};
#pragma unroll
  for (int kt = 0; kt < KT; ++kt) {
    H16 a, x;
    const v8h* w = wa + (size_t)kt * wstep;
    const v8h* q = xa + (size_t)kt * xstep;
    a.h[0] = w[hh];          // A: k = 8*hh + 0..7
    a.h[1] = w[2 + hh];      // A: k = 16 + 8*hh + 0..7
    x.h[0] = q[hh];          // B: k = 8*hh + 0..7
    x.h[1] = q[2 + hh];      // B: k = 16 + 8*hh + 0..7
    acc = WMMA_F16(a.v, x.v, acc);
  }

#pragma unroll
  for (int r = 0; r < 8; ++r) {
    int m = mt * 16 + r + (hh << 3);
    if (m < Mout) {
      float v = acc[r] + (bias ? bias[m] : 0.0f);
      long idx = (long)b * cbs + (long)m * N + nb;
      if (mode == 0)      C[idx] = v;
      else if (mode == 1) C[idx] += v;
      else                C[idx] = R[(long)b * rbs + (long)m * N + nb] + scale * v;
    }
  }
}

// ---------------------------------------------------------------------------
// Dilated 3x3 convs (C_MID=14, dilations 1,2,3), zero padding.
// y[b][i][c][h][w]
// ---------------------------------------------------------------------------
__global__ void dilconv_kernel(const float* __restrict__ z,
                               const float* __restrict__ w,
                               const float* __restrict__ bias,
                               float* __restrict__ y)
{
  int idx = blockIdx.x * blockDim.x + threadIdx.x;
  if (idx >= 2 * 3 * 14 * 64 * 64) return;
  int wp = idx & 63;
  int hp = (idx >> 6) & 63;
  int chan = idx >> 12;            // (b*3+i)*14 + c
  int c  = chan % 14;
  int bi = chan / 14;
  int i  = bi % 3;
  int b  = bi / 3;
  int d  = i + 1;

  float acc = bias[i * 14 + c];
  const float* zb  = z + (size_t)b * 14 * 4096;
  const float* wp9 = w + (size_t)((i * 14 + c) * 14) * 9;
#pragma unroll
  for (int kh = 0; kh < 3; ++kh) {
    int h2 = hp + d * (kh - 1);
    if (h2 < 0 || h2 > 63) continue;
#pragma unroll
    for (int kw = 0; kw < 3; ++kw) {
      int w2 = wp + d * (kw - 1);
      if (w2 < 0 || w2 > 63) continue;
      for (int ci = 0; ci < 14; ++ci)
        acc += wp9[ci * 9 + kh * 3 + kw] * zb[ci * 4096 + h2 * 64 + w2];
    }
  }
  y[idx] = acc;
}

// ---------------------------------------------------------------------------
// Rotation products, written DIRECTLY in packed f16 layout:
//   featsp[b][kt=ch/32][n][ch%32] = y[b][i][c][.] * rot_r(z)[b][c][.]
// ch in [0,256): ch>=252 writes zero padding. j=ch/14, i=j/6, r=j%6, c=ch%14.
// ---------------------------------------------------------------------------
__global__ void mulrot_pack_kernel(const float* __restrict__ y,
                                   const float* __restrict__ z,
                                   _Float16* __restrict__ featsp)
{
  int idx = blockIdx.x * blockDim.x + threadIdx.x;
  if (idx >= 2 * 256 * 64 * 64) return;
  int n  = idx & 4095;
  int wv = idx & 63;
  int hv = (idx >> 6) & 63;
  int ch = (idx >> 12) & 255;
  int b  = idx >> 20;

  float val = 0.0f;
  if (ch < 252) {
    int c = ch % 14;
    int j = ch / 14;
    int i = j / 6;
    int r = j % 6;
    float yv = y[(size_t)((b * 3 + i) * 14 + c) * 4096 + hv * 64 + wv];
    int h2, w2;
    switch (r) {
      case 0:  h2 = hv;      w2 = wv;      break;   // identity
      case 1:  h2 = hv;      w2 = 63 - wv; break;   // flip -1
      case 2:  h2 = 63 - hv; w2 = wv;      break;   // flip -2
      case 3:  h2 = wv;      w2 = 63 - hv; break;   // rot90 k=1
      case 4:  h2 = 63 - hv; w2 = 63 - wv; break;   // rot90 k=2
      default: h2 = 63 - wv; w2 = hv;      break;   // rot90 k=3
    }
    val = yv * z[((size_t)b * 14 + c) * 4096 + h2 * 64 + w2];
  }
  featsp[((size_t)b * 8 * 4096 + (size_t)(ch >> 5) * 4096 + n) * 32 + (ch & 31)] =
      (_Float16)val;
}

// ---------------------------------------------------------------------------
// Transposed flash attention on packed f16 q/k/v.
//   S[m_key][n_query] = K^T Q / 8 ; online softmax over keys with per-lane
//   (per-query-column) running stats; O[d][q] = V * P.
// grid = (64, NH, B), block = 128 (4 waves; one wave = 16 query columns)
// ---------------------------------------------------------------------------
__global__ __launch_bounds__(128) void flashattn_kernel(
    const _Float16* __restrict__ Qp,   // [bh][n][64]
    const _Float16* __restrict__ Kp,   // [bh][m][64]
    const _Float16* __restrict__ Vp,   // [bh][d][4096]
    float* __restrict__ out)           // [b][256][4096]
{
  const int n = 4096;
  const int lane  = threadIdx.x & 31;
  const int wave  = threadIdx.x >> 5;
  const int head  = blockIdx.y;
  const int b     = blockIdx.z;
  const int qtile = blockIdx.x * 4 + wave;
  const int l16 = lane & 15;
  const int hh  = lane >> 4;
  const int nq  = qtile * 16 + l16;

  const size_t bh = (size_t)(b * 4 + head);
  const _Float16* Qb = Qp + bh * n * 64;
  const _Float16* Kb = Kp + bh * n * 64;
  const _Float16* Vb = Vp + bh * 64 * n;

  // Q fragments (B layout), loop-invariant. d-split t=0: 0..31, t=1: 32..63
  const v8h* qv = (const v8h*)(Qb + (size_t)nq * 64);
  H16 qf[2];
#pragma unroll
  for (int t = 0; t < 2; ++t) {
    qf[t].h[0] = qv[t * 4 + hh];
    qf[t].h[1] = qv[t * 4 + 2 + hh];
  }

  float mrun = -1e30f, lrun = 0.0f;
  v8f O0 = {}, O1 = {}, O2 = {}, O3 = {};

  for (int kb = 0; kb < n; kb += 32) {
    // speculative prefetch of next key/value block (unconditional, no branch)
    __builtin_prefetch(Kb + (size_t)(kb + 32 + l16) * 64, 0, 1);
    __builtin_prefetch(Vb + (size_t)l16 * n + kb + 32, 0, 1);

    // ---- S = K^T Q for two 16-key subtiles ----
    v8f s0 = {}, s1 = {};
    const v8h* kv0 = (const v8h*)(Kb + (size_t)(kb + l16) * 64);
    const v8h* kv1 = (const v8h*)(Kb + (size_t)(kb + 16 + l16) * 64);
#pragma unroll
    for (int t = 0; t < 2; ++t) {
      H16 kf;
      kf.h[0] = kv0[t * 4 + hh];
      kf.h[1] = kv0[t * 4 + 2 + hh];
      s0 = WMMA_F16(kf.v, qf[t].v, s0);
    }
#pragma unroll
    for (int t = 0; t < 2; ++t) {
      H16 kf;
      kf.h[0] = kv1[t * 4 + hh];
      kf.h[1] = kv1[t * 4 + 2 + hh];
      s1 = WMMA_F16(kf.v, qf[t].v, s1);
    }

    // ---- online softmax (stats per query column = per lane) ----
    float p0[8], p1[8];
    float sm = -1e30f;
#pragma unroll
    for (int r = 0; r < 8; ++r) {
      p0[r] = s0[r] * 0.125f;
      p1[r] = s1[r] * 0.125f;
      sm = fmaxf(sm, fmaxf(p0[r], p1[r]));
    }
    sm = fmaxf(sm, __shfl_xor(sm, 16, 32));
    float mnew  = fmaxf(mrun, sm);
    float alpha = __expf(mrun - mnew);
    float ls = 0.0f;
#pragma unroll
    for (int r = 0; r < 8; ++r) {
      p0[r] = __expf(p0[r] - mnew);
      p1[r] = __expf(p1[r] - mnew);
      ls += p0[r] + p1[r];
    }
    ls += __shfl_xor(ls, 16, 32);
    lrun = lrun * alpha + ls;
    mrun = mnew;
#pragma unroll
    for (int r = 0; r < 8; ++r) { O0[r] *= alpha; O1[r] *= alpha; O2[r] *= alpha; O3[r] *= alpha; }

    // ---- redistribute P (C layout) into a B fragment: one half-exchange ----
    float kvv[16];
#pragma unroll
    for (int r = 0; r < 8; ++r) {
      float e0 = __shfl_xor(p0[r], 16, 32);
      float e1 = __shfl_xor(p1[r], 16, 32);
      kvv[r]     = hh ? e1    : p0[r];
      kvv[8 + r] = hh ? p1[r] : e0;
    }
    H16 pf;
#pragma unroll
    for (int e = 0; e < 16; ++e) pf.v[e] = (_Float16)kvv[e];

    // ---- O[d][q] += V * P  (4 d-tiles of 16) ----
#pragma unroll
    for (int dt = 0; dt < 4; ++dt) {
      const v8h* vv = (const v8h*)(Vb + (size_t)(dt * 16 + l16) * n + kb);
      H16 vf;
      vf.h[0] = vv[2 * hh];
      vf.h[1] = vv[2 * hh + 1];
      if (dt == 0)      O0 = WMMA_F16(vf.v, pf.v, O0);
      else if (dt == 1) O1 = WMMA_F16(vf.v, pf.v, O1);
      else if (dt == 2) O2 = WMMA_F16(vf.v, pf.v, O2);
      else              O3 = WMMA_F16(vf.v, pf.v, O3);
    }
  }

  const float inv = 1.0f / lrun;
  float* ob = out + (size_t)b * 256 * n + (size_t)(head * 64) * n;
#pragma unroll
  for (int r = 0; r < 8; ++r) {
    int drow = r + (hh << 3);
    ob[(size_t)(drow)      * n + nq] = O0[r] * inv;
    ob[(size_t)(16 + drow) * n + nq] = O1[r] * inv;
    ob[(size_t)(32 + drow) * n + nq] = O2[r] * inv;
    ob[(size_t)(48 + drow) * n + nq] = O3[r] * inv;
  }
}

// ---------------------------------------------------------------------------
// Host-side pipeline
// ---------------------------------------------------------------------------
extern "C" void kernel_launch(void* const* d_in, const int* in_sizes, int n_in,
                              void* d_out, int out_size, void* d_ws, size_t ws_size,
                              hipStream_t stream) {
  const float* x      = (const float*)d_in[0];
  const float* w_tok  = (const float*)d_in[1];
  const float* b_tok  = (const float*)d_in[2];
  const float* w_red  = (const float*)d_in[3];
  const float* b_red  = (const float*)d_in[4];
  const float* w_dil  = (const float*)d_in[5];
  const float* b_dil  = (const float*)d_in[6];
  const float* w_fuse = (const float*)d_in[7];
  const float* b_fuse = (const float*)d_in[8];
  const float* w_qkv  = (const float*)d_in[9];
  const float* w_proj = (const float*)d_in[10];
  const float* w_feat = (const float*)d_in[11];
  const float* b_feat = (const float*)d_in[12];
  float* out = (float*)d_out;

  const long NPIX = 4096;
  char* wsb = (char*)d_ws;
  size_t off = 0;
  auto alloc = [&](size_t bytes) {
    char* p = wsb + off;
    off += (bytes + 255) & ~(size_t)255;
    return p;
  };

  // f32 buffers
  float* token  = (float*)alloc(2L * 256 * NPIX * 4);
  float* zbuf   = (float*)alloc(2L * 14  * NPIX * 4);
  float* ybuf   = (float*)alloc(2L * 42  * NPIX * 4);
  float* qkv    = (float*)alloc(2L * 768 * NPIX * 4);
  float* attout = (float*)alloc(2L * 256 * NPIX * 4);
  // packed f16 buffers
  _Float16* xp      = (_Float16*)alloc(2L * 256 * NPIX * 2);   // [b][8][n][32]
  _Float16* tokp    = (_Float16*)alloc(2L * 256 * NPIX * 2);
  _Float16* featsp  = (_Float16*)alloc(2L * 256 * NPIX * 2);
  _Float16* attoutp = (_Float16*)alloc(2L * 256 * NPIX * 2);
  _Float16* qpk     = (_Float16*)alloc(2L * 4 * NPIX * 64 * 2);
  _Float16* kpk     = (_Float16*)alloc(2L * 4 * NPIX * 64 * 2);
  _Float16* vpk     = (_Float16*)alloc(2L * 4 * NPIX * 64 * 2);
  // packed weights
  _Float16* wtokp  = (_Float16*)alloc(8L * 256 * 32 * 2);
  _Float16* wredp  = (_Float16*)alloc(8L * 16  * 32 * 2);
  _Float16* wfusep = (_Float16*)alloc(8L * 256 * 32 * 2);
  _Float16* wqkvp  = (_Float16*)alloc(8L * 768 * 32 * 2);
  _Float16* wprojp = (_Float16*)alloc(8L * 256 * 32 * 2);
  _Float16* wfeatp = (_Float16*)alloc(8L * 256 * 32 * 2);

  dim3 blk(256);
  const long PBS = 8L * NPIX * 32;   // packed activation batch stride (halves)

  // ---- pack weights (once per call; cheap) ----
  pack_w_kernel<<<(8 * 256 * 32 + 255) / 256, blk, 0, stream>>>(w_tok, 256, 1, 256, 256, 256, 8, wtokp);
  pack_w_kernel<<<(8 * 16 * 32 + 255) / 256, blk, 0, stream>>>(w_red, 256, 1, 14, 256, 16, 8, wredp);
  pack_w_kernel<<<(8 * 256 * 32 + 255) / 256, blk, 0, stream>>>(w_fuse, 252, 1, 256, 252, 256, 8, wfusep);
  pack_w_kernel<<<(8 * 768 * 32 + 255) / 256, blk, 0, stream>>>(w_qkv, 256, 1, 768, 256, 768, 8, wqkvp);
  pack_w_kernel<<<(8 * 256 * 32 + 255) / 256, blk, 0, stream>>>(w_proj, 256, 1, 256, 256, 256, 8, wprojp);
  pack_w_kernel<<<(8 * 256 * 32 + 255) / 256, blk, 0, stream>>>(w_feat, 1, 256, 256, 256, 256, 8, wfeatp);  // transposed

  const int PACKX_BLOCKS = (8 * 4096 * 32 + 255) / 256;

  // 1) pack x ; token = w_tok . x + b_tok
  pack_x_kernel<<<dim3(PACKX_BLOCKS, 1, 2), blk, 0, stream>>>(x, 256L * NPIX, 256, 4096, 8, xp, PBS);
  gemm_wmma_packed<8><<<dim3(512, 1, 2), blk, 0, stream>>>(
      wtokp, xp, PBS, b_tok, token, 256L * NPIX, nullptr, 0, 0.0f, 256, 256, 4096, 0);

  // 2) pack token ; z = w_red . token + b_red   (Mp=16, Mout=14)
  pack_x_kernel<<<dim3(PACKX_BLOCKS, 1, 2), blk, 0, stream>>>(token, 256L * NPIX, 256, 4096, 8, tokp, PBS);
  gemm_wmma_packed<8><<<dim3(32, 1, 2), blk, 0, stream>>>(
      wredp, tokp, PBS, b_red, zbuf, 14L * NPIX, nullptr, 0, 0.0f, 16, 14, 4096, 0);

  // 3) dilated convs ; 4) rotation products -> packed feats
  dilconv_kernel<<<(2 * 3 * 14 * 4096 + 255) / 256, blk, 0, stream>>>(zbuf, w_dil, b_dil, ybuf);
  mulrot_pack_kernel<<<(2 * 256 * 4096 + 255) / 256, blk, 0, stream>>>(ybuf, zbuf, featsp);

  // 5) token += w_fuse . feats + b_fuse
  gemm_wmma_packed<8><<<dim3(512, 1, 2), blk, 0, stream>>>(
      wfusep, featsp, PBS, b_fuse, token, 256L * NPIX, nullptr, 0, 0.0f, 256, 256, 4096, 1);

  // 6) repack token ; qkv = w_qkv . token
  pack_x_kernel<<<dim3(PACKX_BLOCKS, 1, 2), blk, 0, stream>>>(token, 256L * NPIX, 256, 4096, 8, tokp, PBS);
  gemm_wmma_packed<8><<<dim3(1536, 1, 2), blk, 0, stream>>>(
      wqkvp, tokp, PBS, nullptr, qkv, 768L * NPIX, nullptr, 0, 0.0f, 768, 768, 4096, 0);

  // 7) pack q/k/v ; flash attention
  pack_qkv_kernel<<<(2 * 4 * 4096 * 64 + 255) / 256, blk, 0, stream>>>(qkv, qpk, kpk, vpk);
  flashattn_kernel<<<dim3(64, 4, 2), dim3(128), 0, stream>>>(qpk, kpk, vpk, attout);

  // 8) pack attout ; token += w_proj . attout
  pack_x_kernel<<<dim3(PACKX_BLOCKS, 1, 2), blk, 0, stream>>>(attout, 256L * NPIX, 256, 4096, 8, attoutp, PBS);
  gemm_wmma_packed<8><<<dim3(512, 1, 2), blk, 0, stream>>>(
      wprojp, attoutp, PBS, nullptr, token, 256L * NPIX, nullptr, 0, 0.0f, 256, 256, 4096, 1);

  // 9) repack token ; out = x + 0.2 * (w_feat^T . token + b_feat)
  pack_x_kernel<<<dim3(PACKX_BLOCKS, 1, 2), blk, 0, stream>>>(token, 256L * NPIX, 256, 4096, 8, tokp, PBS);
  gemm_wmma_packed<8><<<dim3(512, 1, 2), blk, 0, stream>>>(
      wfeatp, tokp, PBS, b_feat, out, 256L * NPIX, x, 256L * NPIX, 0.2f, 256, 256, 4096, 2);
}
